// Transformer_59863254171867
// MI455X (gfx1250) — matrix operand order repaired and checked by reading, not compiled, not run on hardware
//
#include <hip/hip_runtime.h>

// ---------------- problem constants ----------------
#define TOK    8192      // B*N tokens
#define DIMM   1024
#define SEQ    512
#define NBATCH 16
#define NHEAD  16
#define DHEAD  64
#define MLPD   4096
#define NDEPTH 5

// ---------------- CDNA5 WMMA types ----------------
typedef __attribute__((ext_vector_type(16))) __bf16          v16bf;
typedef __attribute__((ext_vector_type(8)))  unsigned short  v8us;
typedef __attribute__((ext_vector_type(8)))  float           v8f;
typedef __attribute__((ext_vector_type(4)))  unsigned int    u32x4;
typedef __attribute__((ext_vector_type(8)))  int             i32x8;
typedef __attribute__((ext_vector_type(4)))  int             i32x4;

union frag_u { v8us h[2]; v16bf f; };

#if __has_builtin(__builtin_amdgcn_tensor_load_to_lds)
#define USE_TDM 1
#else
#define USE_TDM 0
#endif

__device__ __forceinline__ unsigned short f2bf_bits(float f) {
    unsigned int u = __builtin_bit_cast(unsigned int, f);
    u += 0x7FFFu + ((u >> 16) & 1u);          // round-to-nearest-even
    return (unsigned short)(u >> 16);
}

// raw v_max_num_f32 (skip fmaxf's per-operand canonicalize/quiet ops)
__device__ __forceinline__ float fast_max(float a, float b) {
    float d;
    asm("v_max_num_f32 %0, %1, %2" : "=v"(d) : "v"(a), "v"(b));
    return d;
}

// DPP ROW_XMASK butterfly step (VALU cross-lane, no LDS, no waits).
template <int MASK>
__device__ __forceinline__ float dpp_xor(float x) {
    return __builtin_bit_cast(float,
        __builtin_amdgcn_update_dpp(0, __builtin_bit_cast(int, x),
                                    0x160 | MASK, 0xf, 0xf, true));
}
__device__ __forceinline__ float rowmax16(float x) {
    x = fast_max(x, dpp_xor<1>(x));
    x = fast_max(x, dpp_xor<2>(x));
    x = fast_max(x, dpp_xor<4>(x));
    x = fast_max(x, dpp_xor<8>(x));
    return x;
}
__device__ __forceinline__ float rowsum16(float x) {
    x += dpp_xor<1>(x);
    x += dpp_xor<2>(x);
    x += dpp_xor<4>(x);
    x += dpp_xor<8>(x);
    return x;
}

// ---------------- fp32 -> bf16 converters ----------------
__global__ void cvt_f32_bf16_kernel(const float* __restrict__ s,
                                    unsigned short* __restrict__ d, long long n) {
    long long i = (long long)blockIdx.x * blockDim.x + threadIdx.x;
    long long st = (long long)gridDim.x * blockDim.x;
    for (; i < n; i += st) d[i] = f2bf_bits(s[i]);
}
__global__ void cvt_stride2_kernel(const float* __restrict__ s,
                                   unsigned short* __restrict__ d, long long n, int sel) {
    long long i = (long long)blockIdx.x * blockDim.x + threadIdx.x;
    long long st = (long long)gridDim.x * blockDim.x;
    for (; i < n; i += st) d[i] = f2bf_bits(s[2 * i + sel]);
}

// ---------------- LayerNorm (fp32 in, bf16 out), one row per block --------
__global__ __launch_bounds__(256)
void layernorm_bf16_kernel(const float* __restrict__ x,
                           const float* __restrict__ g,
                           const float* __restrict__ bb,
                           unsigned short* __restrict__ out) {
    __shared__ float red[256];
    const int row = blockIdx.x, tid = threadIdx.x;
    const float4 xv = *(const float4*)(x + (long long)row * DIMM + tid * 4);
    red[tid] = xv.x + xv.y + xv.z + xv.w; __syncthreads();
    for (int st = 128; st > 0; st >>= 1) { if (tid < st) red[tid] += red[tid + st]; __syncthreads(); }
    const float mu = red[0] * (1.0f / DIMM);
    __syncthreads();
    const float d0 = xv.x - mu, d1 = xv.y - mu, d2 = xv.z - mu, d3 = xv.w - mu;
    red[tid] = d0 * d0 + d1 * d1 + d2 * d2 + d3 * d3; __syncthreads();
    for (int st = 128; st > 0; st >>= 1) { if (tid < st) red[tid] += red[tid + st]; __syncthreads(); }
    const float rinv = rsqrtf(red[0] * (1.0f / DIMM) + 1e-5f);
    const int c = tid * 4;
    unsigned short* o = out + (long long)row * DIMM + c;
    o[0] = f2bf_bits(d0 * rinv * g[c + 0] + bb[c + 0]);
    o[1] = f2bf_bits(d1 * rinv * g[c + 1] + bb[c + 1]);
    o[2] = f2bf_bits(d2 * rinv * g[c + 2] + bb[c + 2]);
    o[3] = f2bf_bits(d3 * rinv * g[c + 3] + bb[c + 3]);
}

// ---------------- generic bf16 WMMA GEMM with fused epilogue --------------
// block tile 128x128, BK=32, 8 waves (4x2), each wave 32x64 (2x4 WMMA tiles).
// A tile: TDM (tensor_load_to_lds) DMA into double-buffered LDS, pipelined
//         with s_wait_tensorcnt. 80B pitch reproduced via D# pad_interval=16
//         DWORDs, pad_amount=4 DWORDs.
// B tile: reg-pipelined global loads, TRANSPOSED in LDS -> b128 fragments.
#define BM 128
#define BN 128
#define BK 32

template <int BIASM, int GELU, int RES, int OUTB, int SCL>
__global__ __launch_bounds__(256)
void gemm_bf16_kernel(const unsigned short* __restrict__ A,
                      const unsigned short* __restrict__ Bm,
                      float* __restrict__ outF,
                      unsigned short* __restrict__ outB,
                      const float* __restrict__ bias,
                      const float* __restrict__ residual,
                      int M, int N, int K,
                      long long sA, long long sB, long long sC, long long sR) {
    __shared__ unsigned short As[2][BM][BK + 8];   // double-buffered, pitch 80B
    __shared__ unsigned short Bt[BN][BK + 8];      // [n][k] transposed tile

    const int z = blockIdx.z;
    const unsigned short* Ab = A  + (long long)z * sA;
    const unsigned short* Bb = Bm + (long long)z * sB;
    const long long cOff = (long long)z * sC;

    const int tid  = threadIdx.x;
    const int wave = tid >> 5, lane = tid & 31;
    const int hf   = lane >> 4, ln = lane & 15;
    const int wm   = wave & 3,  wn = wave >> 2;
    const int blockM = blockIdx.y * BM, blockN = blockIdx.x * BN;

    const int bRow = tid >> 5, bC4 = (tid & 31) * 4;        // B: 4 x uint2 (coalesced)
#if !USE_TDM
    const int aRow = tid >> 1, aC4 = (tid & 1) * 16;        // A fallback: 2 x uint4
    uint4 aReg[2];
#endif

    const v8f z8 = {0.f, 0.f, 0.f, 0.f, 0.f, 0.f, 0.f, 0.f};
    v8f acc[2][4] = {{z8, z8, z8, z8}, {z8, z8, z8, z8}};

    uint2 bReg[4];
    auto fetchB = [&](int k0) {
        #pragma unroll
        for (int i = 0; i < 4; ++i)
            bReg[i] = *(const uint2*)(Bb + (long long)(k0 + bRow + 8 * i) * N + blockN + bC4);
    };
    auto stageB = [&]() {
        #pragma unroll
        for (int i = 0; i < 4; ++i) {            // scatter-transpose B: [n][k]
            const int kk = bRow + 8 * i;
            Bt[bC4 + 0][kk] = (unsigned short)(bReg[i].x & 0xFFFF);
            Bt[bC4 + 1][kk] = (unsigned short)(bReg[i].x >> 16);
            Bt[bC4 + 2][kk] = (unsigned short)(bReg[i].y & 0xFFFF);
            Bt[bC4 + 3][kk] = (unsigned short)(bReg[i].y >> 16);
        }
    };

#if USE_TDM
    // D# (ISA 8.3-8.6): count=1, type=2, data_size=2B, tile 32x128,
    // pad every 16 DWORDs (64B row) with 4 DWORDs (16B) -> 80B LDS pitch.
    auto tdm_issue = [&](int kt, int buf) {
        const unsigned long long ga =
            (unsigned long long)(Ab + (long long)blockM * K + kt * BK);
        const unsigned ldsa =
            (unsigned)(unsigned long long)(const void*)&As[buf][0][0];
        u32x4 g0;
        g0.x = 1u;                                          // count=1
        g0.y = ldsa;                                        // lds_addr
        g0.z = (unsigned)ga;                                // global_addr lo
        g0.w = (unsigned)(ga >> 32) | (2u << 30);           // addr hi | type=2
        i32x8 g1;
        g1[0] = (int)((1u << 16) | (1u << 20) | (3u << 22) | (3u << 25));
        g1[1] = (int)((unsigned)K << 16);                   // tensor_dim0 lo
        g1[2] = (int)(((unsigned)M & 0xFFFFu) << 16);       // dim0 hi=0, tensor_dim1 lo
        g1[3] = (int)(32u << 16);                           // dim1 hi=0, tile_dim0=32
        g1[4] = 128;                                        // tile_dim1=128, tile_dim2=0
        g1[5] = K;                                          // tensor_dim0_stride lo
        g1[6] = 0; g1[7] = 0;
        const i32x4 zz4 = {0, 0, 0, 0};
        const i32x8 zz8 = {0, 0, 0, 0, 0, 0, 0, 0};
        __builtin_amdgcn_tensor_load_to_lds(g0, g1, zz4, zz4, zz8, 0);
    };
#endif

    const int nk = K / BK;
    fetchB(0);
#if USE_TDM
    if (wave == 0) tdm_issue(0, 0);
#else
    #pragma unroll
    for (int i = 0; i < 2; ++i)
        aReg[i] = *(const uint4*)(Ab + (long long)(blockM + aRow) * K + aC4 + 8 * i);
#endif

    for (int kt = 0; kt < nk; ++kt) {
        stageB();
#if USE_TDM
        const int abuf = kt & 1;
        if (wave == 0) {
            if (kt + 1 < nk) {
                tdm_issue(kt + 1, (kt + 1) & 1);            // prefetch next A tile
                __builtin_amdgcn_s_wait_tensorcnt(1);       // current tile complete
            } else {
                __builtin_amdgcn_s_wait_tensorcnt(0);
            }
        }
#else
        const int abuf = 0;
        #pragma unroll
        for (int i = 0; i < 2; ++i)
            *(uint4*)&As[0][aRow][aC4 + 8 * i] = aReg[i];
#endif
        __syncthreads();
        if (kt + 1 < nk) {
            fetchB((kt + 1) * BK);                          // overlap with math
#if !USE_TDM
            #pragma unroll
            for (int i = 0; i < 2; ++i)
                aReg[i] = *(const uint4*)(Ab + (long long)(blockM + aRow) * K +
                                          (kt + 1) * BK + aC4 + 8 * i);
#endif
        }

        frag_u afr[2], bfr[4];
        #pragma unroll
        for (int i = 0; i < 2; ++i) {
            const int row = wm * 32 + i * 16 + ln;
            afr[i].h[0] = *(const v8us*)&As[abuf][row][8 * hf];
            afr[i].h[1] = *(const v8us*)&As[abuf][row][16 + 8 * hf];
        }
        #pragma unroll
        for (int j = 0; j < 4; ++j) {
            const int col = wn * 64 + j * 16 + ln;
            bfr[j].h[0] = *(const v8us*)&Bt[col][8 * hf];
            bfr[j].h[1] = *(const v8us*)&Bt[col][16 + 8 * hf];
        }
        #pragma unroll
        for (int i = 0; i < 2; ++i)
            #pragma unroll
            for (int j = 0; j < 4; ++j)
                acc[i][j] = __builtin_amdgcn_wmma_f32_16x16x32_bf16(
                    false, afr[i].f, false, bfr[j].f, (short)0, acc[i][j], false, false);
        __syncthreads();
    }

    // epilogue (straight-line; C layout: VGPR r -> row r+8*hf, lane&15 -> col)
    #pragma unroll
    for (int j = 0; j < 4; ++j) {
        const int col = blockN + wn * 64 + j * 16 + ln;
        float bcol = 0.f;
        if constexpr (BIASM == 1) bcol = bias[col];
        #pragma unroll
        for (int i = 0; i < 2; ++i) {
            const int rbase = blockM + wm * 32 + i * 16 + 8 * hf;
            #pragma unroll
            for (int r = 0; r < 8; ++r) {
                const int gr = rbase + r;
                float v = acc[i][j][r];
                if constexpr (BIASM == 1) v += bcol;
                if constexpr (BIASM == 2) v += bias[gr];
                if constexpr (GELU)       v = 0.5f * v * (1.0f + erff(v * 0.70710678118654752f));
                if constexpr (RES)        v += residual[(long long)z * sR + (long long)gr * N + col];
                if constexpr (SCL)        v *= 0.125f;
                const long long ci = cOff + (long long)gr * N + col;
                if constexpr (OUTB) outB[ci] = f2bf_bits(v);
                else                outF[ci] = v;
            }
        }
    }
}

// ---------------- fused flash attention ------------------------------------
__global__ __launch_bounds__(256)
void attention_kernel(const unsigned short* __restrict__ q,
                      const unsigned short* __restrict__ k,
                      const unsigned short* __restrict__ v,
                      unsigned short* __restrict__ o,
                      int ldq, int ldkv, int ldo) {
    __shared__ unsigned short Ks[32][72];        // [key][d]   pitch 144B
    __shared__ unsigned short Vt[64][40];        // [d][key]   pitch 80B
    __shared__ unsigned short pst[8][16][40];    // wave-private P staging

    const int tid = threadIdx.x, wave = tid >> 5, lane = tid & 31;
    const int hf = lane >> 4, ln = lane & 15;
    const int gw = blockIdx.x * 8 + wave;
    const int qt = gw & 31;
    const int h  = (gw >> 5) & 15;
    const int b  = gw >> 9;
    const int q0 = qt * 16;
    const long long row0 = (long long)b * SEQ;

    const int sKey = tid >> 4, sC4 = (tid & 15) * 4;

    frag_u aq[2];
    #pragma unroll
    for (int c = 0; c < 2; ++c) {
        const unsigned short* base = q + (row0 + q0 + ln) * ldq + h * DHEAD + 32 * c;
        aq[c].h[0] = *(const v8us*)(base + 8 * hf);
        aq[c].h[1] = *(const v8us*)(base + 16 + 8 * hf);
    }

    const v8f z8 = {0.f, 0.f, 0.f, 0.f, 0.f, 0.f, 0.f, 0.f};
    v8f accO[4] = {z8, z8, z8, z8};
    float mrun[8], lrun[8];
    #pragma unroll
    for (int r = 0; r < 8; ++r) { mrun[r] = -3.0e38f; lrun[r] = 0.f; }

    for (int kb = 0; kb < SEQ / 32; ++kb) {
        const int key0 = kb * 32;
        #pragma unroll
        for (int i = 0; i < 2; ++i) {
            const int key = sKey + 16 * i;
            const uint2 kv2 = *(const uint2*)(k + (row0 + key0 + key) * ldkv + h * DHEAD + sC4);
            *(uint2*)&Ks[key][sC4] = kv2;
            const uint2 vv2 = *(const uint2*)(v + (row0 + key0 + key) * ldkv + h * DHEAD + sC4);
            Vt[sC4 + 0][key] = (unsigned short)(vv2.x & 0xFFFF);
            Vt[sC4 + 1][key] = (unsigned short)(vv2.x >> 16);
            Vt[sC4 + 2][key] = (unsigned short)(vv2.y & 0xFFFF);
            Vt[sC4 + 3][key] = (unsigned short)(vv2.y >> 16);
        }
        __syncthreads();

        // S = Q @ K^T (chains interleaved to cover WMMA->VALU hazards)
        v8f s[2] = {z8, z8};
        #pragma unroll
        for (int c = 0; c < 2; ++c) {
            #pragma unroll
            for (int jt = 0; jt < 2; ++jt) {
                const int key = jt * 16 + ln;
                frag_u bk;
                bk.h[0] = *(const v8us*)&Ks[key][32 * c + 8 * hf];
                bk.h[1] = *(const v8us*)&Ks[key][32 * c + 16 + 8 * hf];
                s[jt] = __builtin_amdgcn_wmma_f32_16x16x32_bf16(
                    false, aq[c].f, false, bk.f, (short)0, s[jt], false, false);
            }
        }

        // online softmax: DPP row reductions, raw v_max_num
        float alpha[8];
        #pragma unroll
        for (int r = 0; r < 8; ++r) {
            const float rm = rowmax16(fast_max(s[0][r], s[1][r]));
            const float mn = fast_max(mrun[r], rm);
            const float al = __expf(mrun[r] - mn);
            const float p0 = __expf(s[0][r] - mn), p1 = __expf(s[1][r] - mn);
            s[0][r] = p0; s[1][r] = p1;
            lrun[r] = lrun[r] * al + rowsum16(p0 + p1);
            mrun[r] = mn;
            alpha[r] = al;
        }
        #pragma unroll
        for (int j = 0; j < 4; ++j)
            #pragma unroll
            for (int r = 0; r < 8; ++r) accO[j][r] *= alpha[r];

        // P: C-layout -> LDS -> A-fragment
        #pragma unroll
        for (int r = 0; r < 8; ++r) {
            pst[wave][8 * hf + r][ln]      = f2bf_bits(s[0][r]);
            pst[wave][8 * hf + r][16 + ln] = f2bf_bits(s[1][r]);
        }
        __builtin_amdgcn_wave_barrier();
        frag_u pa;
        pa.h[0] = *(const v8us*)&pst[wave][ln][8 * hf];
        pa.h[1] = *(const v8us*)&pst[wave][ln][16 + 8 * hf];

        // O += P @ V
        #pragma unroll
        for (int j = 0; j < 4; ++j) {
            frag_u bv;
            bv.h[0] = *(const v8us*)&Vt[16 * j + ln][8 * hf];
            bv.h[1] = *(const v8us*)&Vt[16 * j + ln][16 + 8 * hf];
            accO[j] = __builtin_amdgcn_wmma_f32_16x16x32_bf16(
                false, pa.f, false, bv.f, (short)0, accO[j], false, false);
        }
        __syncthreads();
    }

    float rinv[8];
    #pragma unroll
    for (int r = 0; r < 8; ++r) rinv[r] = 1.0f / lrun[r];
    #pragma unroll
    for (int j = 0; j < 4; ++j)
        #pragma unroll
        for (int r = 0; r < 8; ++r) {
            const int row = q0 + r + 8 * hf;
            o[(row0 + row) * ldo + h * DHEAD + 16 * j + ln] = f2bf_bits(accO[j][r] * rinv[r]);
        }
}

// ---------------- host orchestration ----------------
extern "C" void kernel_launch(void* const* d_in, const int* in_sizes, int n_in,
                              void* d_out, int out_size, void* d_ws, size_t ws_size,
                              hipStream_t stream) {
    (void)in_sizes; (void)n_in; (void)out_size; (void)ws_size;
    const float* x_in   = (const float*)d_in[0];
    const float* ln1_g  = (const float*)d_in[1];
    const float* ln1_b  = (const float*)d_in[2];
    const float* Wq     = (const float*)d_in[3];
    const float* Wkv    = (const float*)d_in[4];
    const float* Wo     = (const float*)d_in[5];
    const float* bo     = (const float*)d_in[6];
    const float* ln2_g  = (const float*)d_in[7];
    const float* ln2_b  = (const float*)d_in[8];
    const float* W1     = (const float*)d_in[9];
    const float* b1     = (const float*)d_in[10];
    const float* W2     = (const float*)d_in[11];
    const float* b2     = (const float*)d_in[12];
    const float* skip_w = (const float*)d_in[13];
    const float* skip_b = (const float*)d_in[14];
    float* out = (float*)d_out;

    char* base = (char*)d_ws; size_t off = 0;
    auto alloc = [&](size_t bytes) -> void* {
        void* p = base + off; off = (off + bytes + 255) & ~(size_t)255; return p;
    };
    const size_t fX = (size_t)TOK * DIMM * sizeof(float);
    float* xA = (float*)alloc(fX);
    float* xB = (float*)alloc(fX);
    float* lo[3] = {(float*)alloc(fX), (float*)alloc(fX), (float*)alloc(fX)};
    unsigned short* xn  = (unsigned short*)alloc((size_t)TOK * DIMM * 2);
    unsigned short* qb  = (unsigned short*)alloc((size_t)TOK * DIMM * 2);
    unsigned short* kvb = (unsigned short*)alloc((size_t)TOK * 2 * DIMM * 2);
    unsigned short* ob  = (unsigned short*)alloc((size_t)TOK * DIMM * 2);
    unsigned short* hb  = (unsigned short*)alloc((size_t)TOK * MLPD * 2);
    unsigned short* wb  = (unsigned short*)alloc((size_t)MLPD * DIMM * 2);

    auto cvt = [&](const float* s, unsigned short* d, long long n) {
        cvt_f32_bf16_kernel<<<1024, 256, 0, stream>>>(s, d, n);
    };

    (void)hipMemcpyAsync(xA, x_in, fX, hipMemcpyDeviceToDevice, stream);
    float* x = xA; float* xo = xB;

    const long long sBD = (long long)SEQ * DIMM;
    for (int nl = 0; nl < NDEPTH; ++nl) {
        (void)hipMemcpyAsync(lo[nl % 3], x, fX, hipMemcpyDeviceToDevice, stream);
        if (nl > 1) {
            const float* sw = skip_w + (size_t)(nl - 2) * SEQ * SEQ * 2;
            cvt_stride2_kernel<<<1024, 256, 0, stream>>>(sw, wb, (long long)SEQ * SEQ, 0);
            cvt_stride2_kernel<<<1024, 256, 0, stream>>>(sw, wb + SEQ * SEQ, (long long)SEQ * SEQ, 1);
            cvt(x, xn, (long long)TOK * DIMM);
            cvt(lo[(nl - 2) % 3], qb, (long long)TOK * DIMM);
            gemm_bf16_kernel<2, 0, 0, 0, 0><<<dim3(DIMM / BN, SEQ / BM, NBATCH), 256, 0, stream>>>(
                wb, xn, out, nullptr, skip_b + (size_t)(nl - 2) * SEQ, nullptr,
                SEQ, DIMM, SEQ, 0, sBD, sBD, 0);
            gemm_bf16_kernel<0, 0, 1, 0, 0><<<dim3(DIMM / BN, SEQ / BM, NBATCH), 256, 0, stream>>>(
                wb + SEQ * SEQ, qb, xo, nullptr, nullptr, out,
                SEQ, DIMM, SEQ, 0, sBD, sBD, sBD);
            { float* t = x; x = xo; xo = t; }
        }
        layernorm_bf16_kernel<<<TOK, 256, 0, stream>>>(x, ln1_g + (size_t)nl * DIMM,
                                                       ln1_b + (size_t)nl * DIMM, xn);
        cvt(Wq + (size_t)nl * DIMM * DIMM, wb, (long long)DIMM * DIMM);
        gemm_bf16_kernel<0, 0, 0, 1, 1><<<dim3(DIMM / BN, TOK / BM, 1), 256, 0, stream>>>(
            xn, wb, nullptr, qb, nullptr, nullptr, TOK, DIMM, DIMM, 0, 0, 0, 0);
        cvt(Wkv + (size_t)nl * DIMM * 2 * DIMM, wb, (long long)DIMM * 2 * DIMM);
        gemm_bf16_kernel<0, 0, 0, 1, 0><<<dim3(2 * DIMM / BN, TOK / BM, 1), 256, 0, stream>>>(
            xn, wb, nullptr, kvb, nullptr, nullptr, TOK, 2 * DIMM, DIMM, 0, 0, 0, 0);
        attention_kernel<<<1024, 256, 0, stream>>>(qb, kvb, kvb + DIMM, ob,
                                                   DIMM, 2 * DIMM, DIMM);
        cvt(Wo + (size_t)nl * DIMM * DIMM, wb, (long long)DIMM * DIMM);
        gemm_bf16_kernel<1, 0, 1, 0, 0><<<dim3(DIMM / BN, TOK / BM, 1), 256, 0, stream>>>(
            ob, wb, xo, nullptr, bo + (size_t)nl * DIMM, x, TOK, DIMM, DIMM, 0, 0, 0, 0);
        { float* t = x; x = xo; xo = t; }
        layernorm_bf16_kernel<<<TOK, 256, 0, stream>>>(x, ln2_g + (size_t)nl * DIMM,
                                                       ln2_b + (size_t)nl * DIMM, xn);
        cvt(W1 + (size_t)nl * DIMM * MLPD, wb, (long long)DIMM * MLPD);
        gemm_bf16_kernel<1, 1, 0, 1, 0><<<dim3(MLPD / BN, TOK / BM, 1), 256, 0, stream>>>(
            xn, wb, nullptr, hb, b1 + (size_t)nl * MLPD, nullptr, TOK, MLPD, DIMM, 0, 0, 0, 0);
        cvt(W2 + (size_t)nl * MLPD * DIMM, wb, (long long)MLPD * DIMM);
        gemm_bf16_kernel<1, 0, 1, 0, 0><<<dim3(DIMM / BN, TOK / BM, 1), 256, 0, stream>>>(
            hb, wb, xo, nullptr, b2 + (size_t)nl * DIMM, x, TOK, DIMM, MLPD, 0, 0, 0, 0);
        { float* t = x; x = xo; xo = t; }
    }
    (void)hipMemcpyAsync(out, x, fX, hipMemcpyDeviceToDevice, stream);
}
